// CRF_88064009437461
// MI455X (gfx1250) — compile-verified, matrix operand order checked by reading, not run on hardware
//
#include <hip/hip_runtime.h>

// CRF negative log-likelihood for B=256, S=2048, T=48 on gfx1250 (MI455X).
//
// Denominator (log-partition) is computed in the *scaled linear domain*:
//   p_t = normalize( (M^T p_{t-1}) .* exp(e_t) ),  M = exp(transitions)
// so the per-step recurrence is a 48x48 x 48x16 f16 WMMA GEMM per 16-batch
// tile, with batch mapped to the WMMA N dimension so the f32 D-tile layout
// coincides lane-for-lane with the f16 B-fragment layout of the next step
// (no LDS, no cross-lane shuffles on the critical path).

typedef __attribute__((ext_vector_type(16))) _Float16 v16h;
typedef __attribute__((ext_vector_type(8)))  float    v8f;

constexpr int BB = 256;   // batch
constexpr int SS = 2048;  // sequence length
constexpr int TT = 48;    // tags/states

// ---------------------------------------------------------------------------
// Kernel 1: gold-path score (numerator). One thread per batch row.
// ---------------------------------------------------------------------------
__global__ __launch_bounds__(256) void crf_numerator_kernel(
    const float* __restrict__ em, const int* __restrict__ tags,
    const int* __restrict__ mask, const float* __restrict__ trans,
    const float* __restrict__ startT, const float* __restrict__ endT,
    float* __restrict__ num) {
  const int b = blockIdx.x * blockDim.x + threadIdx.x;
  if (b >= BB) return;
  const int*   tg = tags + (size_t)b * SS;
  const int*   mk = mask + (size_t)b * SS;
  const float* e  = em   + (size_t)b * SS * TT;

  int   prev  = tg[0];
  float score = startT[prev] + e[prev];
  for (int t = 1; t < SS; ++t) {
    const int   cur = tg[t];
    const int   m   = mk[t];
    const float mf  = (float)m;
    score += (trans[prev * TT + cur] + e[(size_t)t * TT + cur]) * mf;
    prev = (m != 0) ? cur : prev;
  }
  num[b] = score + endT[prev];
}

// ---------------------------------------------------------------------------
// Kernel 2: forward algorithm (denominator) + final out = denom - num.
// One wave (32 lanes) per 16-batch tile; grid = 16 blocks.
//   lane = {col = lane&15 (batch column / WMMA N), hi = lane>>4}
//   D/C f32 tile layout: element (M = r + 8*hi, N = col) in VGPR r.
//   A f16 16x32 layout:  lane M = col, half q -> K = (q<8?q:q+8) + 8*hi.
//   B f16 32x16 layout:  lane N = col, half q -> K = (q<8?q:q+8) + 8*hi.
// ---------------------------------------------------------------------------
__global__ __launch_bounds__(32) void crf_forward_kernel(
    const float* __restrict__ em, const int* __restrict__ mask,
    const float* __restrict__ trans, const float* __restrict__ startT,
    const float* __restrict__ endT, const float* __restrict__ num,
    float* __restrict__ out) {
  const int lane = threadIdx.x;
  const int col  = lane & 15;
  const int hi   = lane >> 4;
  const int b    = blockIdx.x * 16 + col;

  // --- constant A fragments: A[m][kb] holds A[M][K] = exp(trans[K][M]),
  //     M = 16m + col (output state), K = input state (0 for K >= 48).
  v16h A[3][2];
#pragma unroll
  for (int m = 0; m < 3; ++m) {
    const int M = 16 * m + col;
#pragma unroll
    for (int kb = 0; kb < 2; ++kb) {
#pragma unroll
      for (int q = 0; q < 16; ++q) {
        const int K = kb * 32 + (q < 8 ? q : q + 8) + hi * 8;
        const float v = (K < TT) ? __expf(trans[K * TT + M]) : 0.0f;
        A[m][kb][q] = (_Float16)v;
      }
    }
  }

  const float* erow = em   + (size_t)b * SS * TT;
  const int*   mrow = mask + (size_t)b * SS;

  // --- t = 0: p[j] = exp(start[j] + e0[j]); lane holds j = 16m + 8*hi + r.
  float P[3][8];
  float cs = 0.0f;
#pragma unroll
  for (int m = 0; m < 3; ++m) {
    const int j0 = 16 * m + 8 * hi;
    const float4 a4 = *(const float4*)(erow + j0);
    const float4 b4 = *(const float4*)(erow + j0 + 4);
    const float ev[8] = {a4.x, a4.y, a4.z, a4.w, b4.x, b4.y, b4.z, b4.w};
#pragma unroll
    for (int r = 0; r < 8; ++r) {
      const float qv = __expf(startT[j0 + r] + ev[r]);
      P[m][r] = qv;
      cs += qv;
    }
  }
  cs += __shfl_xor(cs, 16, 32);
  float logAcc;
  {
    const float rs = __builtin_amdgcn_rcpf(cs);
    logAcc = -__logf(rs);  // exact bookkeeping: rcp error cancels in alpha
#pragma unroll
    for (int m = 0; m < 3; ++m)
#pragma unroll
      for (int r = 0; r < 8; ++r) P[m][r] *= rs;
  }

  // --- B fragments for step t come straight from P (lane-local repack).
  v16h B0, B1;
#pragma unroll
  for (int q = 0; q < 8; ++q) {
    B0[q]     = (_Float16)P[0][q];
    B0[q + 8] = (_Float16)P[1][q];
    B1[q]     = (_Float16)P[2][q];
    B1[q + 8] = (_Float16)0.0f;
  }

  // --- software pipeline: emissions/mask for t+1 loaded during step t.
  float eN[3][8];
  int   mN = mrow[1];
#pragma unroll
  for (int m = 0; m < 3; ++m) {
    const float* p0 = erow + TT + 16 * m + 8 * hi;
    const float4 a4 = *(const float4*)(p0);
    const float4 b4 = *(const float4*)(p0 + 4);
    eN[m][0] = a4.x; eN[m][1] = a4.y; eN[m][2] = a4.z; eN[m][3] = a4.w;
    eN[m][4] = b4.x; eN[m][5] = b4.y; eN[m][6] = b4.z; eN[m][7] = b4.w;
  }

  for (int t = 1; t < SS; ++t) {
    float eC[3][8];
#pragma unroll
    for (int m = 0; m < 3; ++m)
#pragma unroll
      for (int r = 0; r < 8; ++r) eC[m][r] = eN[m][r];
    const int mC = mN;

    // deep prefetch: pull emission lines ~16 steps ahead into cache
    if (t + 16 < SS) {
#pragma unroll
      for (int m = 0; m < 3; ++m)
        __builtin_prefetch(erow + (size_t)(t + 16) * TT + 16 * m + 8 * hi, 0, 3);
    }
    // 1-step register prefetch (independent of the recurrence chain)
    if (t + 1 < SS) {
      mN = mrow[t + 1];
#pragma unroll
      for (int m = 0; m < 3; ++m) {
        const float* p0 = erow + (size_t)(t + 1) * TT + 16 * m + 8 * hi;
        const float4 a4 = *(const float4*)(p0);
        const float4 b4 = *(const float4*)(p0 + 4);
        eN[m][0] = a4.x; eN[m][1] = a4.y; eN[m][2] = a4.z; eN[m][3] = a4.w;
        eN[m][4] = b4.x; eN[m][5] = b4.y; eN[m][6] = b4.z; eN[m][7] = b4.w;
      }
    }

    // D(48x16) = exp(trans)^T (48x48, padded to 64) x P (48x16, padded)
    v8f D0 = {}, D1 = {}, D2 = {};
    D0 = __builtin_amdgcn_wmma_f32_16x16x32_f16(false, A[0][0], false, B0,
                                                (short)0, D0, false, false);
    D0 = __builtin_amdgcn_wmma_f32_16x16x32_f16(false, A[0][1], false, B1,
                                                (short)0, D0, false, false);
    D1 = __builtin_amdgcn_wmma_f32_16x16x32_f16(false, A[1][0], false, B0,
                                                (short)0, D1, false, false);
    D1 = __builtin_amdgcn_wmma_f32_16x16x32_f16(false, A[1][1], false, B1,
                                                (short)0, D1, false, false);
    D2 = __builtin_amdgcn_wmma_f32_16x16x32_f16(false, A[2][0], false, B0,
                                                (short)0, D2, false, false);
    D2 = __builtin_amdgcn_wmma_f32_16x16x32_f16(false, A[2][1], false, B1,
                                                (short)0, D2, false, false);

    // q = D .* exp(e_t); column sum for renormalization
    float Q[3][8];
    float cs2 = 0.0f;
#pragma unroll
    for (int r = 0; r < 8; ++r) {
      Q[0][r] = D0[r] * __expf(eC[0][r]); cs2 += Q[0][r];
      Q[1][r] = D1[r] * __expf(eC[1][r]); cs2 += Q[1][r];
      Q[2][r] = D2[r] * __expf(eC[2][r]); cs2 += Q[2][r];
    }
    cs2 += __shfl_xor(cs2, 16, 32);  // lanes l and l+16 share batch column

    const bool  upd = (mC != 0);     // mask: lane-local selects, EXEC untouched
    const float rs  = __builtin_amdgcn_rcpf(cs2);
    const float nl  = -__logf(rs);
    logAcc += upd ? nl : 0.0f;
#pragma unroll
    for (int m = 0; m < 3; ++m)
#pragma unroll
      for (int r = 0; r < 8; ++r)
        P[m][r] = upd ? Q[m][r] * rs : P[m][r];

    // repack next B fragments (pure per-lane cvt, no LDS / no shuffles)
#pragma unroll
    for (int q = 0; q < 8; ++q) {
      B0[q]     = (_Float16)P[0][q];
      B0[q + 8] = (_Float16)P[1][q];
      B1[q]     = (_Float16)P[2][q];
      B1[q + 8] = (_Float16)0.0f;
    }
  }

  // --- denominator = logAcc + log( sum_j P[j] * exp(end[j]) )
  float fs = 0.0f;
#pragma unroll
  for (int m = 0; m < 3; ++m) {
    const int j0 = 16 * m + 8 * hi;
#pragma unroll
    for (int r = 0; r < 8; ++r) fs += P[m][r] * __expf(endT[j0 + r]);
  }
  fs += __shfl_xor(fs, 16, 32);
  const float denom = logAcc + __logf(fs);

  if (hi == 0) out[b] = denom - num[b];
}

// ---------------------------------------------------------------------------
// Launch: numerator -> d_ws, then forward (reads d_ws) -> d_out. Same stream,
// so ordering is guaranteed; both kernels are graph-capture safe.
// ---------------------------------------------------------------------------
extern "C" void kernel_launch(void* const* d_in, const int* in_sizes, int n_in,
                              void* d_out, int out_size, void* d_ws,
                              size_t ws_size, hipStream_t stream) {
  (void)in_sizes; (void)n_in; (void)out_size; (void)ws_size;
  const float* em     = (const float*)d_in[0];
  const int*   tags   = (const int*)d_in[1];
  const int*   mask   = (const int*)d_in[2];
  const float* trans  = (const float*)d_in[3];
  const float* startT = (const float*)d_in[4];
  const float* endT   = (const float*)d_in[5];
  float*       outp   = (float*)d_out;
  float*       num    = (float*)d_ws;  // 256 floats of scratch

  crf_numerator_kernel<<<dim3(1), dim3(256), 0, stream>>>(
      em, tags, mask, trans, startT, endT, num);
  crf_forward_kernel<<<dim3(BB / 16), dim3(32), 0, stream>>>(
      em, mask, trans, startT, endT, num, outp);
}
